// MultiHeadAttention_60507499266616
// MI455X (gfx1250) — matrix operand order repaired and checked by reading, not compiled, not run on hardware
//
#include <hip/hip_runtime.h>

// ---------------------------------------------------------------------------
// Types
// ---------------------------------------------------------------------------
typedef __bf16 bf16;
typedef __attribute__((ext_vector_type(8)))  __bf16 v8bf;
typedef __attribute__((ext_vector_type(16))) __bf16 v16bf;
typedef __attribute__((ext_vector_type(8)))  float  v8f;
typedef __attribute__((ext_vector_type(4)))  float  v4f;
typedef __attribute__((ext_vector_type(4)))  unsigned int u32x4;
typedef __attribute__((ext_vector_type(8)))  unsigned int u32x8;

__device__ __forceinline__ bf16 f2bf(float f) {
  return (bf16)f;   // native v_cvt on gfx1250 (RNE)
}

__device__ __forceinline__ v8f wmma_bf16(v16bf a, v16bf b, v8f c) {
  // D = A(16x32 bf16) x B(32x16 bf16) + C(16x16 f32)
  return __builtin_amdgcn_wmma_f32_16x16x32_bf16(
      /*neg_a=*/false, a, /*neg_b=*/false, b,
      /*c_mod=*/(short)0, c, /*reuse_a=*/false, /*reuse_b=*/false);
}

// ---------------------------------------------------------------------------
// Async global -> LDS copy (CDNA5, ASYNCcnt-tracked). Each active lane copies
// 16 bytes from its global address to its LDS address with no VGPR transit.
// LDS byte offset = low 32 bits of the generic pointer (flat->LDS aperture
// mapping).
// ---------------------------------------------------------------------------
__device__ __forceinline__ void async_copy_b128(const void* gsrc, void* ldst) {
  asm volatile("global_load_async_to_lds_b128 %0, %1, off"
               :: "v"((unsigned)(unsigned long long)ldst),
                  "v"((unsigned long long)gsrc)
               : "memory");
}
__device__ __forceinline__ void wait_async0() {
  asm volatile("s_wait_asynccnt 0x0" ::: "memory");
}

// ---------------------------------------------------------------------------
// Tensor Data Mover: one wave issues a single DMA of a 2-D tile
// (tile_d1 rows x tile_d0 elems, 2-byte elements) from global memory into
// LDS, with optional per-row LDS padding. Descriptor layout per CDNA5 ISA
// ch.8 (D# group0 = 4 SGPRs, group1 = 8 SGPRs; groups 2/3 omitted for 2D).
// Tracked by TENSORcnt; drain with s_wait_tensorcnt.
// ---------------------------------------------------------------------------
__device__ __forceinline__ void tdm_load_2d_b16(
    unsigned lds_byte_addr, const void* gaddr,
    unsigned tile_d0_elems, unsigned tile_d1_rows,
    unsigned tensor_d0_elems, unsigned d0_stride_elems,
    unsigned pad_en, unsigned pad_interval_code, unsigned pad_amount_code) {
  const unsigned long long ga = (unsigned long long)gaddr;
  u32x4 g0;
  g0[0] = 1u;                                          // count=1, user mode
  g0[1] = lds_byte_addr;                               // lds_addr [63:32]
  g0[2] = (unsigned)(ga & 0xFFFFFFFFu);                // global_addr[31:0]
  g0[3] = (unsigned)((ga >> 32) & 0x1FFFFFFu)          // global_addr[56:32]
        | (2u << 30);                                  // type = 2 ("image")
  u32x8 g1;
  g1[0] = (1u << 16)                                   // data_size = 2 bytes
        | (pad_en << 20)
        | (pad_interval_code << 22)
        | (pad_amount_code << 25);
  g1[1] = (tensor_d0_elems & 0xFFFFu) << 16;           // tensor_dim0[15:0]
  g1[2] = ((tensor_d0_elems >> 16) & 0xFFFFu)          // tensor_dim0[31:16]
        | (0xFFFFu << 16);                             // tensor_dim1[15:0] (big)
  g1[3] = 0x0000FFFFu                                  // tensor_dim1[31:16]
        | ((tile_d0_elems & 0xFFFFu) << 16);           // tile_dim0
  g1[4] = (tile_d1_rows & 0xFFFFu);                    // tile_dim1 (tile_dim2=0)
  g1[5] = d0_stride_elems;                             // tensor_dim0_stride lo
  g1[6] = 0u;                                          // stride hi / dim1_stride
  g1[7] = 0u;
  asm volatile("tensor_load_to_lds %0, %1" :: "s"(g0), "s"(g1) : "memory");
}

// ---------------------------------------------------------------------------
// Fragment loaders (wave32 layouts per CDNA5 ISA 7.12.2)
//
// A (16x32, 16-bit): lanes 0-15 -> M=lane, elements [0..7]=K0..7, [8..15]=K16..23
//                    lanes 16-31 -> M=lane-16, elements [0..7]=K8..15, [8..15]=K24..31
// B (32x16, 16-bit): lanes 0-15 -> N=lane, elements [0..7]=K0..7, [8..15]=K16..23
//                    lanes 16-31 -> N=lane-16, elements [0..7]=K8..15, [8..15]=K24..31
// LDS for A-frag is row-major [m][k]; LDS for B-frag is [n][k].
// ---------------------------------------------------------------------------
template<int LD>
__device__ __forceinline__ v16bf load_frag_a(const bf16* smem, int row0, int k0,
                                             int l16, int lhalf) {
  const bf16* base = smem + (row0 + l16) * LD + k0 + lhalf * 8;
  v8bf lo = *(const v8bf*)base;          // K: k0 + lhalf*8 .. +7
  v8bf hi = *(const v8bf*)(base + 16);   // K: k0 + 16 + lhalf*8 .. +7
  v16bf r;
#pragma unroll
  for (int i = 0; i < 8; ++i) { r[i] = lo[i]; r[i + 8] = hi[i]; }
  return r;
}

template<int LD>
__device__ __forceinline__ v16bf load_frag_b(const bf16* smem, int n0, int k0,
                                             int l16, int lhalf) {
  const bf16* base = smem + (n0 + l16) * LD + k0 + lhalf * 8;
  v8bf lo = *(const v8bf*)base;          // K: k0 + lhalf*8 .. +7
  v8bf hi = *(const v8bf*)(base + 16);   // K: k0 + 16 + lhalf*8 .. +7
  v16bf r;
#pragma unroll
  for (int i = 0; i < 8; ++i) { r[i] = lo[i]; r[i + 8] = hi[i]; }
  return r;
}

// ---------------------------------------------------------------------------
// Output store helpers
// ---------------------------------------------------------------------------
__device__ __forceinline__ void store_out(float* p, float v) { *p = v; }
__device__ __forceinline__ void store_out(bf16* p, float v) { *p = f2bf(v); }

template<typename T> struct is_f32 { static constexpr bool value = false; };
template<> struct is_f32<float>   { static constexpr bool value = true;  };

// ---------------------------------------------------------------------------
// GEMM: Out[M,N] = A[M,K] * W[K,N] + bias[N]
//   A: fp32 or bf16 row-major; W: fp32 row-major; Out: fp32 or bf16.
//   Block tile 128x128, BK=32 (one WMMA K-window). 256 threads = 8 waves,
//   wave grid 4(m) x 2(n), each wave owns 32x64 (2x4 WMMA accumulators).
//   bf16-A path stages its tile with GLOBAL_LOAD_ASYNC_TO_LDS_B128.
// ---------------------------------------------------------------------------
template<typename AT, typename OT>
__global__ void __launch_bounds__(256)
gemm_bias_wmma(const AT* __restrict__ A, const float* __restrict__ W,
               const float* __restrict__ bias, OT* __restrict__ Out,
               int M, int N, int K) {
  __shared__ bf16 As[128 * 40];   // [m][k], padded stride 40
  __shared__ bf16 Bs[128 * 40];   // [n][k], padded stride 40

  const int tid  = threadIdx.x;
  const int wave = tid >> 5;
  const int lane = tid & 31;
  const int l16  = lane & 15;
  const int lhalf = lane >> 4;
  const int wm = wave >> 1;       // 0..3  -> m offset wm*32
  const int wn = wave & 1;        // 0..1  -> n offset wn*64
  const int mbase = blockIdx.y * 128;
  const int nbase = blockIdx.x * 128;

  v8f acc[2][4] = {};

  // cooperative-load coordinates
  const int arow = tid >> 1;            // 0..127
  const int acol = (tid & 1) * 16;      // 0 or 16
  const int bk   = (tid >> 5) * 4;      // 0..28
  const int bn   = (tid & 31) * 4;      // 0..124

  for (int kt = 0; kt < K; kt += 32) {
    // ---- stage A tile (128 x 32) into LDS as bf16 ----
    {
      const AT* ap = A + (size_t)(mbase + arow) * K + kt + acol;
      if constexpr (is_f32<AT>::value) {
        bf16 tmp[16];
#pragma unroll
        for (int c = 0; c < 4; ++c) {
          v4f f = *(const v4f*)(ap + c * 4);
#pragma unroll
          for (int e = 0; e < 4; ++e) tmp[c * 4 + e] = f2bf(f[e]);
        }
        v8bf s0, s1;
#pragma unroll
        for (int e = 0; e < 8; ++e) { s0[e] = tmp[e]; s1[e] = tmp[e + 8]; }
        *(v8bf*)&As[arow * 40 + acol]     = s0;
        *(v8bf*)&As[arow * 40 + acol + 8] = s1;
      } else {
        // layout-preserving bf16 copy: async DMA straight into LDS
        async_copy_b128(ap,     &As[arow * 40 + acol]);
        async_copy_b128(ap + 8, &As[arow * 40 + acol + 8]);
      }
      if (kt + 32 < K) __builtin_prefetch(ap + 32, 0, 1);   // global_prefetch_b8
    }
    // ---- stage W tile (32 x 128) transposed into LDS [n][k] as bf16 ----
    {
#pragma unroll
      for (int r = 0; r < 4; ++r) {
        const float* wp = W + (size_t)(kt + bk + r) * N + nbase + bn;
        v4f f = *(const v4f*)wp;
#pragma unroll
        for (int c = 0; c < 4; ++c) Bs[(bn + c) * 40 + bk + r] = f2bf(f[c]);
      }
      if (kt + 32 < K)
        __builtin_prefetch(W + (size_t)(kt + 32 + bk) * N + nbase + bn, 0, 1);
    }
    if constexpr (!is_f32<AT>::value) wait_async0();
    __syncthreads();

    // ---- compute: 8 WMMAs per wave per K-tile ----
    v16bf a0 = load_frag_a<40>(As, wm * 32,      0, l16, lhalf);
    v16bf a1 = load_frag_a<40>(As, wm * 32 + 16, 0, l16, lhalf);
#pragma unroll
    for (int ni = 0; ni < 4; ++ni) {
      v16bf b = load_frag_b<40>(Bs, wn * 64 + ni * 16, 0, l16, lhalf);
      acc[0][ni] = wmma_bf16(a0, b, acc[0][ni]);
      acc[1][ni] = wmma_bf16(a1, b, acc[1][ni]);
    }
    __syncthreads();
  }

  // ---- epilogue: bias + store (D layout: elem r -> m = +lhalf*8+r, n = +l16) ----
#pragma unroll
  for (int mi = 0; mi < 2; ++mi) {
#pragma unroll
    for (int ni = 0; ni < 4; ++ni) {
      const int ncol = nbase + wn * 64 + ni * 16 + l16;
      const float bv = bias[ncol];
#pragma unroll
      for (int r = 0; r < 8; ++r) {
        const int mrow = mbase + wm * 32 + mi * 16 + lhalf * 8 + r;
        store_out(&Out[(size_t)mrow * N + ncol], acc[mi][ni][r] + bv);
      }
    }
  }
}

// ---------------------------------------------------------------------------
// Flash attention forward per (b, h, 64-row q tile).
//   Q/K/V: bf16 [B*S, H*64] (head h at column offset h*64).
//   Tq = Tk = 64, DH = 64. 256 threads = 8 waves; wave grid 4(m) x 2(n):
//   wave owns 16 q-rows x 32 cols (cols = keys for QK^T, = d for PV).
//   Q tile staged by per-lane async DMA; K tile staged by one TDM
//   tensor_load_to_lds per iteration (wave 0, TENSORcnt); V needs an
//   element transpose so it goes through VGPRs.
// ---------------------------------------------------------------------------
__global__ void __launch_bounds__(256)
attn_fwd_wmma(const bf16* __restrict__ Q, const bf16* __restrict__ Kx,
              const bf16* __restrict__ V, const unsigned char* __restrict__ mask,
              bf16* __restrict__ O, int S, int Hn) {
  __shared__ bf16  Qs[64 * 72];    // [q][d]
  __shared__ bf16  Ks[64 * 72];    // [key][d]  == B-frag layout for QK^T
  __shared__ bf16  Vts[64 * 72];   // [d][key]  == B-frag layout for P*V
  __shared__ float Ss[64 * 68];    // fp32 scores
  __shared__ bf16  Ps[64 * 72];    // bf16 probabilities (A-frag layout)
  __shared__ float alpha_s[64];
  __shared__ float linv_s[64];

  const int tid  = threadIdx.x;
  const int wave = tid >> 5;
  const int lane = tid & 31;
  const int l16  = lane & 15;
  const int lhalf = lane >> 4;
  const int wm = wave >> 1;        // q-row base wm*16
  const int wn = wave & 1;         // col base wn*32
  const int q0 = blockIdx.x * 64;
  const int h  = blockIdx.y;
  const int b  = blockIdx.z;
  const size_t rowstride = (size_t)Hn * 64;                    // = D
  const size_t qbase = ((size_t)b * S + q0) * rowstride + (size_t)h * 64;

  const int lr = tid >> 2;          // 0..63 (tile row for cooperative loads)
  const int lc = (tid & 3) * 16;    // 0,16,32,48 (tile col)

  // ---- issue async load of the Q tile (waited in the first iteration) ----
  {
    const bf16* src = Q + qbase + (size_t)lr * rowstride + lc;
    async_copy_b128(src,     &Qs[lr * 72 + lc]);
    async_copy_b128(src + 8, &Qs[lr * 72 + lc + 8]);
  }

  v8f oacc[2] = {};
  float mrun = -3.0e38f, lrun = 0.0f;   // per-row state, valid in tid < 64

  for (int kv = 0; kv < S; kv += 64) {
    __syncthreads();   // prior iteration done reading Ks/Vts/Ps
    // ---- K tile [key][d]: one TDM DMA for the whole 64x64 tile (wave 0).
    //      LDS row pad: 128B rows (32 DW -> interval code 4) + 16B pad
    //      (4 DW -> amount code 3) => 144B stride == Ks stride of 72 elems.
    if (wave == 0) {
      tdm_load_2d_b16((unsigned)(unsigned long long)&Ks[0],
                      Kx + ((size_t)b * S + kv) * rowstride + (size_t)h * 64,
                      /*tile_d0=*/64, /*tile_d1=*/64,
                      /*tensor_d0=*/64, /*d0_stride=*/(unsigned)rowstride,
                      /*pad_en=*/1u, /*interval=*/4u, /*amount=*/3u);
    }
    // ---- V tile transposed into [d][key] via VGPRs ----
    {
      const size_t gro = ((size_t)b * S + kv + lr) * rowstride + (size_t)h * 64 + lc;
      v8bf y0 = *(const v8bf*)(V + gro);
      v8bf y1 = *(const v8bf*)(V + gro + 8);
#pragma unroll
      for (int i = 0; i < 8; ++i) {
        Vts[(lc + i) * 72 + lr]     = y0[i];
        Vts[(lc + 8 + i) * 72 + lr] = y1[i];
      }
    }
    wait_async0();                          // Q tile (first iteration)
    __builtin_amdgcn_s_wait_tensorcnt(0);   // K tile DMA complete
    __syncthreads();

    // ---- S = (Q K^T) * 1/sqrt(64) ----
    {
      v8f sacc[2] = {};
#pragma unroll
      for (int ks = 0; ks < 64; ks += 32) {
        v16bf a = load_frag_a<72>(Qs, wm * 16, ks, l16, lhalf);
#pragma unroll
        for (int ni = 0; ni < 2; ++ni) {
          v16bf bb = load_frag_b<72>(Ks, wn * 32 + ni * 16, ks, l16, lhalf);
          sacc[ni] = wmma_bf16(a, bb, sacc[ni]);
        }
      }
#pragma unroll
      for (int ni = 0; ni < 2; ++ni) {
        const int n = wn * 32 + ni * 16 + l16;
#pragma unroll
        for (int r = 0; r < 8; ++r) {
          const int m = wm * 16 + lhalf * 8 + r;
          Ss[m * 68 + n] = sacc[ni][r] * 0.125f;
        }
      }
    }
    __syncthreads();

    // ---- online softmax: thread tid<64 owns q-row tid ----
    if (tid < 64) {
      const int r = tid;
      const unsigned char* mrow = mask + ((size_t)b * S + q0 + r) * S + kv;
      float mx = mrun;
#pragma unroll 8
      for (int j = 0; j < 64; ++j) {
        float s = mrow[j] ? Ss[r * 68 + j] : -3.0e38f;
        Ss[r * 68 + j] = s;
        mx = fmaxf(mx, s);
      }
      const float alpha = __expf(mrun - mx);
      float sum = 0.0f;
#pragma unroll 8
      for (int j = 0; j < 64; ++j) {
        float p = __expf(Ss[r * 68 + j] - mx);
        sum += p;
        Ps[r * 72 + j] = f2bf(p);
      }
      lrun = lrun * alpha + sum;
      mrun = mx;
      alpha_s[r] = alpha;
    }
    __syncthreads();

    // ---- O = O*alpha + P V ----
    {
      float a8[8];
#pragma unroll
      for (int r = 0; r < 8; ++r) a8[r] = alpha_s[wm * 16 + lhalf * 8 + r];
#pragma unroll
      for (int ni = 0; ni < 2; ++ni)
#pragma unroll
        for (int r = 0; r < 8; ++r) oacc[ni][r] *= a8[r];
#pragma unroll
      for (int ks = 0; ks < 64; ks += 32) {
        v16bf a = load_frag_a<72>(Ps, wm * 16, ks, l16, lhalf);
#pragma unroll
        for (int ni = 0; ni < 2; ++ni) {
          v16bf bb = load_frag_b<72>(Vts, wn * 32 + ni * 16, ks, l16, lhalf);
          oacc[ni] = wmma_bf16(a, bb, oacc[ni]);
        }
      }
    }
  }

  // ---- finalize: O /= l, write bf16 out in [b, q, h, d] (== [B*S, D]) ----
  if (tid < 64) linv_s[tid] = 1.0f / lrun;
  __syncthreads();
  {
    float li[8];
#pragma unroll
    for (int r = 0; r < 8; ++r) li[r] = linv_s[wm * 16 + lhalf * 8 + r];
#pragma unroll
    for (int ni = 0; ni < 2; ++ni) {
      const int n = wn * 32 + ni * 16 + l16;   // d index
#pragma unroll
      for (int r = 0; r < 8; ++r) {
        const int m = wm * 16 + lhalf * 8 + r; // q row in tile
        O[qbase + (size_t)m * rowstride + n] = f2bf(oacc[ni][r] * li[r]);
      }
    }
  }
}

// ---------------------------------------------------------------------------
// Launch
// ---------------------------------------------------------------------------
extern "C" void kernel_launch(void* const* d_in, const int* in_sizes, int n_in,
                              void* d_out, int out_size, void* d_ws, size_t ws_size,
                              hipStream_t stream) {
  (void)in_sizes; (void)n_in; (void)out_size; (void)ws_size;
  const int Bn = 4, S = 2048, D = 1024, Hn = 16;
  const int M = Bn * S, N = D, K = D;

  const float* query = (const float*)d_in[0];
  const float* key   = (const float*)d_in[1];
  const float* value = (const float*)d_in[2];
  const unsigned char* mask = (const unsigned char*)d_in[3];
  const float* Wq = (const float*)d_in[4];
  const float* bq = (const float*)d_in[5];
  const float* Wk = (const float*)d_in[6];
  const float* bk = (const float*)d_in[7];
  const float* Wv = (const float*)d_in[8];
  const float* bv = (const float*)d_in[9];
  const float* Wo = (const float*)d_in[10];
  const float* bo = (const float*)d_in[11];

  const size_t E = (size_t)M * D;           // 8,388,608 elements
  bf16* Qb = (bf16*)d_ws;                   // each buffer 16.78 MB
  bf16* Kb = Qb + E;
  bf16* Vb = Kb + E;
  bf16* Ab = Vb + E;

  dim3 gg(N / 128, M / 128);                // (8, 64)
  gemm_bias_wmma<float, bf16><<<gg, 256, 0, stream>>>(query, Wq, bq, Qb, M, N, K);
  gemm_bias_wmma<float, bf16><<<gg, 256, 0, stream>>>(key,   Wk, bk, Kb, M, N, K);
  gemm_bias_wmma<float, bf16><<<gg, 256, 0, stream>>>(value, Wv, bv, Vb, M, N, K);

  attn_fwd_wmma<<<dim3(S / 64, Hn, Bn), 256, 0, stream>>>(Qb, Kb, Vb, mask, Ab, S, Hn);

  gemm_bias_wmma<bf16, float><<<gg, 256, 0, stream>>>(Ab, Wo, bo, (float*)d_out, M, N, K);
}
